// GroupedVectorAttention_58789512348489
// MI455X (gfx1250) — compile-verified, matrix operand order checked by reading, not run on hardware
//
#include <hip/hip_runtime.h>
#include <hip/hip_bf16.h>
#include <cmath>

typedef __bf16 bf16;
typedef __attribute__((ext_vector_type(16))) __bf16 v16bf;
typedef __attribute__((ext_vector_type(8)))  __bf16 v8bf;
typedef __attribute__((ext_vector_type(8)))  float  v8f;
typedef __attribute__((ext_vector_type(4)))  int    v4i;

static constexpr int kC      = 128;   // model dim
static constexpr int kDInner = 256;   // 2*C
static constexpr int kDState = 16;
static constexpr int kDtRank = 8;
static constexpr int kDbc    = kDtRank + 2 * kDState; // 40

__device__ __forceinline__ float silu_(float x)     { return x / (1.f + __expf(-x)); }
__device__ __forceinline__ float softplus_(float x) { return x > 20.f ? x : log1pf(__expf(x)); }

__device__ __forceinline__ float wred32(float v) {
#pragma unroll
  for (int m = 16; m >= 1; m >>= 1) v += __shfl_xor(v, m, 32);
  return v;
}

// ---- 16-byte global -> LDS copy, async on gfx1250 when the builtin exists
__device__ __forceinline__ void cp16_g2s(bf16* l, const bf16* g) {
#if __has_builtin(__builtin_amdgcn_global_load_async_to_lds_b128)
  typedef __attribute__((address_space(1))) v4i* gp_t;   // global src
  typedef __attribute__((address_space(3))) v4i* lp_t;   // LDS dst
  __builtin_amdgcn_global_load_async_to_lds_b128(
      (gp_t)(unsigned long long)g, (lp_t)(unsigned long long)l, 0, 0);
#else
  *(v8bf*)l = *(const v8bf*)g;
#endif
}

__device__ __forceinline__ void wait_async_lds() {
#if __has_builtin(__builtin_amdgcn_global_load_async_to_lds_b128)
#if __has_builtin(__builtin_amdgcn_s_wait_asynccnt)
  __builtin_amdgcn_s_wait_asynccnt(0);
#else
  asm volatile("s_wait_asynccnt 0" ::: "memory");
#endif
#endif
}

// ---- weight convert: float (rows x cols, row-major) -> bf16 transposed (cols x rows)
__global__ void k_wT_bf16(const float* __restrict__ src, bf16* __restrict__ dst,
                          int rows, int cols) {
  int i = blockIdx.x * blockDim.x + threadIdx.x;
  int total = rows * cols;
  if (i >= total) return;
  int m = i / rows;        // dst row  (= src col)
  int k = i - m * rows;    // dst col  (= src row)
  dst[(size_t)m * rows + k] = (bf16)src[(size_t)k * cols + m];
}

// ---- rms_norm(feat) -> bf16   (wave per row, 4 cols/lane)
__global__ void k_rms_bf16(const float* __restrict__ feat, const float* __restrict__ w,
                           bf16* __restrict__ out, int N) {
  int n = blockIdx.x * 8 + (threadIdx.x >> 5);
  int lane = threadIdx.x & 31;
  if (n >= N) return;
  const float* r = feat + (size_t)n * kC;
  float v[4]; float s = 0.f;
#pragma unroll
  for (int i = 0; i < 4; i++) { v[i] = r[lane + 32 * i]; s += v[i] * v[i]; }
  s = wred32(s);
  float inv = rsqrtf(s / (float)kC + 1e-5f);
#pragma unroll
  for (int i = 0; i < 4; i++) {
    int c = lane + 32 * i;
    out[(size_t)n * kC + c] = (bf16)(v[i] * inv * w[c]);
  }
}

// ---- gather + gauss-weighted sum + layer_norm -> cat[:,128:256] (bf16)
__global__ void k_combined(const float* __restrict__ feat, const int* __restrict__ ridx,
                           const float* __restrict__ gauss,
                           const float* __restrict__ g, const float* __restrict__ b,
                           bf16* __restrict__ catm, int N) {
  int n = blockIdx.x * 8 + (threadIdx.x >> 5);
  int lane = threadIdx.x & 31;
  if (n >= N) return;
  float acc[4] = {0.f, 0.f, 0.f, 0.f};
#pragma unroll 1
  for (int k = 0; k < 16; k++) {
    int j = ridx[n * 16 + k];
    float w = gauss[n * 16 + k];
    const float* fr = feat + (size_t)j * kC;
#pragma unroll
    for (int i = 0; i < 4; i++) acc[i] += w * fr[lane + 32 * i];
  }
  float s = 0.f;
#pragma unroll
  for (int i = 0; i < 4; i++) s += acc[i];
  float mean = wred32(s) / (float)kC;
  float vs = 0.f;
#pragma unroll
  for (int i = 0; i < 4; i++) { float d = acc[i] - mean; vs += d * d; }
  float var = wred32(vs) / (float)kC;
  float inv = rsqrtf(var + 1e-5f);
#pragma unroll
  for (int i = 0; i < 4; i++) {
    int c = lane + 32 * i;
    catm[(size_t)n * (2 * kC) + kC + c] = (bf16)((acc[i] - mean) * inv * g[c] + b[c]);
  }
}

// ---- generic layernorm (f32 in, 128 cols) -> bf16 out at (ldo, coloff), optional relu
__global__ void k_ln_bf16(const float* __restrict__ in, const float* __restrict__ g,
                          const float* __restrict__ b, bf16* __restrict__ out,
                          int ldo, int coloff, int do_relu, int N) {
  int n = blockIdx.x * 8 + (threadIdx.x >> 5);
  int lane = threadIdx.x & 31;
  if (n >= N) return;
  const float* r = in + (size_t)n * kC;
  float v[4]; float s = 0.f;
#pragma unroll
  for (int i = 0; i < 4; i++) { v[i] = r[lane + 32 * i]; s += v[i]; }
  float mean = wred32(s) / (float)kC;
  float vs = 0.f;
#pragma unroll
  for (int i = 0; i < 4; i++) { float d = v[i] - mean; vs += d * d; }
  float var = wred32(vs) / (float)kC;
  float inv = rsqrtf(var + 1e-5f);
#pragma unroll
  for (int i = 0; i < 4; i++) {
    int c = lane + 32 * i;
    float o = (v[i] - mean) * inv * g[c] + b[c];
    if (do_relu) o = fmaxf(o, 0.f);
    out[(size_t)n * ldo + coloff + c] = (bf16)o;
  }
}

// ---- bf16 fragment gather per ISA 7.12.2: lane L -> row (L&15), K-half (L>>4);
// elements [0..7] = K {base+hsel*8..}, [8..15] = K {base+16+hsel*8..}
__device__ __forceinline__ v16bf frag_ld(const bf16* __restrict__ row_ptr, int k0, int hsel) {
  const v8bf* a = (const v8bf*)(row_ptr + k0 + hsel * 8);
  const v8bf* b = (const v8bf*)(row_ptr + k0 + 16 + hsel * 8);
  v8bf x = *a, y = *b;
  return __builtin_shufflevector(x, y, 0, 1, 2, 3, 4, 5, 6, 7,
                                       8, 9, 10, 11, 12, 13, 14, 15);
}

// ---- WMMA bf16 GEMM:  C[N x M] = A[N x K] * BT[M x K]^T  (+bias, +residual)
// Block = 128 threads = 4 waves; block tile = 16 rows x 64 cols.
// A tile (16xK) and B tile (64xK) staged to LDS via async b128 copies.
// HAS_BIAS / HAS_RES are compile-time so the epilogue is branch-free.
template <int K, bool HAS_BIAS, bool HAS_RES>
__global__ void k_gemm_bf16(const bf16* __restrict__ A, const bf16* __restrict__ BT,
                            float* __restrict__ Cm, int N, int M,
                            const float* __restrict__ bias,
                            const float* __restrict__ residual) {
  constexpr int LDB = K + 8;              // +16B pad: avoid LDS bank conflicts
  constexpr int KC8 = K / 8;              // 16-byte chunks per row
  __shared__ bf16 sA[16 * LDB];
  __shared__ bf16 sB[64 * LDB];

  int tid = threadIdx.x;
  int lane = tid & 31;
  int wave = tid >> 5;
  int tn = blockIdx.x;
  int m0 = blockIdx.y * 64;

  const bf16* gA = A + (size_t)tn * 16 * K;     // 16 contiguous rows
  const bf16* gB = BT + (size_t)m0 * K;         // 64 contiguous rows
#pragma unroll
  for (int l = tid; l < 16 * KC8; l += 128) {
    int row = l / KC8, kc = l - row * KC8;
    cp16_g2s(sA + row * LDB + kc * 8, gA + (size_t)row * K + kc * 8);
  }
#pragma unroll
  for (int l = tid; l < 64 * KC8; l += 128) {
    int row = l / KC8, kc = l - row * KC8;
    cp16_g2s(sB + row * LDB + kc * 8, gB + (size_t)row * K + kc * 8);
  }
  wait_async_lds();
  __syncthreads();

  int hsel = lane >> 4;
  int r16 = lane & 15;
  const bf16* aRow = sA + r16 * LDB;
  const bf16* bRow = sB + (wave * 16 + r16) * LDB;

  v8f acc = {0.f, 0.f, 0.f, 0.f, 0.f, 0.f, 0.f, 0.f};
#pragma unroll
  for (int k0 = 0; k0 < K; k0 += 32) {
    v16bf af = frag_ld(aRow, k0, hsel);
    v16bf bfr = frag_ld(bRow, k0, hsel);
    acc = __builtin_amdgcn_wmma_f32_16x16x32_bf16(
        false, af, false, bfr, (short)0, acc, false, false);
  }

  int colg = m0 + wave * 16 + r16;
  int rbase = tn * 16 + hsel * 8;
  float bv = 0.f;
  if constexpr (HAS_BIAS) bv = bias[colg];
  float* outp = Cm + (size_t)rbase * M + colg;
  const float* resp = HAS_RES ? (residual + (size_t)rbase * M + colg) : nullptr;
#pragma unroll
  for (int r = 0; r < 8; r++) {
    float v = acc[r] + bv;
    if constexpr (HAS_RES) v += resp[(size_t)r * M];
    outp[(size_t)r * M] = v;
  }
}

// ---- causal depthwise conv (window 4) + silu ; x = xz[:, :256]
__global__ void k_conv_silu(const float* __restrict__ xz, const float* __restrict__ cw,
                            const float* __restrict__ cb, float* __restrict__ xc, int N) {
  int i = blockIdx.x * blockDim.x + threadIdx.x;
  if (i >= N * kDInner) return;
  int n = i >> 8, c = i & 255;
  float s = cb[c];
#pragma unroll
  for (int k = 0; k < 4; k++) {
    int t = n - 3 + k;
    if (t >= 0) s += xz[(size_t)t * (2 * kDInner) + c] * cw[c * 4 + k];
  }
  xc[i] = silu_(s);
}

// ---- dbc = xconv @ x_proj_w  (256 -> 40)
__global__ void k_dbc(const float* __restrict__ xc, const float* __restrict__ W,
                      float* __restrict__ dbc, int N) {
  int i = blockIdx.x * blockDim.x + threadIdx.x;
  if (i >= N * kDbc) return;
  int n = i / kDbc, j = i - n * kDbc;
  const float* r = xc + (size_t)n * kDInner;
  float s = 0.f;
  for (int c = 0; c < kDInner; c++) s += r[c] * W[c * kDbc + j];
  dbc[i] = s;
}

// ---- delta = softplus(dbc[:, :8] @ dt_proj_w + dt_proj_b)
__global__ void k_delta(const float* __restrict__ dbc, const float* __restrict__ dtW,
                        const float* __restrict__ dtb, float* __restrict__ delta, int N) {
  int i = blockIdx.x * blockDim.x + threadIdx.x;
  if (i >= N * kDInner) return;
  int n = i >> 8, c = i & 255;
  const float* d = dbc + (size_t)n * kDbc;
  float s = dtb[c];
#pragma unroll
  for (int r = 0; r < kDtRank; r++) s += d[r] * dtW[r * kDInner + c];
  delta[i] = softplus_(s);
}

// ---- selective scan: lanes 0-15 = states of ch c0, lanes 16-31 = ch c0+1
__global__ void k_scan(const float* __restrict__ delta, const float* __restrict__ dbc,
                       const float* __restrict__ xc, const float* __restrict__ A_log,
                       float* __restrict__ y, int N) {
  int lane = threadIdx.x & 31;
  int wave = threadIdx.x >> 5;
  int s = lane & 15;
  int c = blockIdx.x * 16 + wave * 2 + (lane >> 4);
  float a = -__expf(A_log[c * kDState + s]);
  float h = 0.f;
#pragma unroll 1
  for (int t = 0; t < N; ++t) {
    if (t + 32 < N) {   // hide latency of the serial chain
      __builtin_prefetch((const void*)(delta + (size_t)(t + 32) * kDInner + c), 0, 3);
      __builtin_prefetch((const void*)(xc + (size_t)(t + 32) * kDInner + c), 0, 3);
      __builtin_prefetch((const void*)(dbc + (size_t)(t + 32) * kDbc), 0, 3);
    }
    float dt = delta[(size_t)t * kDInner + c];
    float xt = xc[(size_t)t * kDInner + c];
    const float* db = dbc + (size_t)t * kDbc;
    float bt = db[kDtRank + s];
    float ct = db[kDtRank + kDState + s];
    h = __expf(dt * a) * h + dt * bt * xt;
    float p = h * ct;
#pragma unroll
    for (int m = 8; m >= 1; m >>= 1) p += __shfl_xor(p, m, 16);
    if (s == 0) y[(size_t)t * kDInner + c] = p;
  }
}

// ---- ybar = (y + D*xconv) * silu(z) -> bf16 ; z = xz[:, 256:512]
__global__ void k_ypost(const float* __restrict__ y, const float* __restrict__ xc,
                        const float* __restrict__ xz, const float* __restrict__ Dp,
                        bf16* __restrict__ ybar, int N) {
  int i = blockIdx.x * blockDim.x + threadIdx.x;
  if (i >= N * kDInner) return;
  int n = i >> 8, c = i & 255;
  float z = xz[(size_t)n * (2 * kDInner) + kDInner + c];
  ybar[i] = (bf16)((y[i] + Dp[c] * xc[i]) * silu_(z));
}

extern "C" void kernel_launch(void* const* d_in, const int* in_sizes, int n_in,
                              void* d_out, int out_size, void* d_ws, size_t ws_size,
                              hipStream_t stream) {
  const float* feat    = (const float*)d_in[0];
  const int*   ridx    = (const int*)d_in[2];
  const float* gauss   = (const float*)d_in[3];
  const float* in_proj = (const float*)d_in[5];
  const float* conv_w  = (const float*)d_in[6];
  const float* conv_b  = (const float*)d_in[7];
  const float* x_proj  = (const float*)d_in[8];
  const float* dt_w    = (const float*)d_in[9];
  const float* dt_b    = (const float*)d_in[10];
  const float* A_log   = (const float*)d_in[11];
  const float* Dp      = (const float*)d_in[12];
  const float* out_proj= (const float*)d_in[13];
  const float* rms_w   = (const float*)d_in[14];
  const float* ln_g    = (const float*)d_in[15];
  const float* ln_b    = (const float*)d_in[16];
  const float* la_w1   = (const float*)d_in[17];
  const float* la_b1   = (const float*)d_in[18];
  const float* la_ln_g = (const float*)d_in[19];
  const float* la_ln_b = (const float*)d_in[20];
  const float* la_w2   = (const float*)d_in[21];
  const float* la_b2   = (const float*)d_in[22];
  float* out = (float*)d_out;
  (void)n_in; (void)out_size; (void)ws_size;

  const int N = in_sizes[0] / kC;          // 32768
  const int M1 = 2 * kDInner;              // 512

  char* base = (char*)d_ws;
  size_t off = 0;
  auto carve = [&](size_t bytes) -> char* {
    off = (off + 255) & ~(size_t)255;
    char* p = base + off; off += bytes; return p;
  };
  bf16*  xrms  = (bf16*) carve((size_t)N * kC * 2);        // reused as act2
  bf16*  wtin  = (bf16*) carve((size_t)M1 * kC * 2);       // 512x128
  bf16*  wtout = (bf16*) carve((size_t)kC * kDInner * 2);  // 128x256
  bf16*  wtla1 = (bf16*) carve((size_t)kC * kDInner * 2);  // 128x256
  bf16*  wtla2 = (bf16*) carve((size_t)kC * kC * 2);       // 128x128
  float* xz    = (float*)carve((size_t)N * M1 * 4);
  float* xc    = (float*)carve((size_t)N * kDInner * 4);
  float* dbc   = (float*)carve((size_t)N * kDbc * 4);
  float* delta = (float*)carve((size_t)N * kDInner * 4);
  float* ybuf  = (float*)carve((size_t)N * kDInner * 4);
  bf16*  ybar  = (bf16*) carve((size_t)N * kDInner * 2);
  float* fm    = (float*)carve((size_t)N * kC * 4);        // reused as hpre
  bf16*  catm  = (bf16*) carve((size_t)N * kDInner * 2);
  bf16*  act2  = xrms;
  float* hpre  = fm;

  // weights -> bf16 transposed (M x K)
  k_wT_bf16<<<(kC * M1 + 255) / 256, 256, 0, stream>>>(in_proj,  wtin,  kC,      M1);
  k_wT_bf16<<<(kDInner * kC + 255) / 256, 256, 0, stream>>>(out_proj, wtout, kDInner, kC);
  k_wT_bf16<<<(kDInner * kC + 255) / 256, 256, 0, stream>>>(la_w1,    wtla1, kDInner, kC);
  k_wT_bf16<<<(kC * kC + 255) / 256, 256, 0, stream>>>(la_w2,    wtla2, kC,      kC);

  dim3 rowBlk(256); dim3 rowGrd((N + 7) / 8);
  k_rms_bf16 <<<rowGrd, rowBlk, 0, stream>>>(feat, rms_w, xrms, N);
  k_combined <<<rowGrd, rowBlk, 0, stream>>>(feat, ridx, gauss, ln_g, ln_b, catm, N);

  // xz = rms(feat) @ in_proj   (N x 512), K = 128
  k_gemm_bf16<128, false, false><<<dim3(N / 16, M1 / 64), 128, 0, stream>>>(
      xrms, wtin, xz, N, M1, nullptr, nullptr);

  k_conv_silu<<<(N * kDInner + 255) / 256, 256, 0, stream>>>(xz, conv_w, conv_b, xc, N);
  k_dbc      <<<(N * kDbc + 255) / 256, 256, 0, stream>>>(xc, x_proj, dbc, N);
  k_delta    <<<(N * kDInner + 255) / 256, 256, 0, stream>>>(dbc, dt_w, dt_b, delta, N);

  k_scan<<<kDInner / 16, 256, 0, stream>>>(delta, dbc, xc, A_log, ybuf, N);

  k_ypost<<<(N * kDInner + 255) / 256, 256, 0, stream>>>(ybuf, xc, xz, Dp, ybar, N);

  // fm = feat + ybar @ out_proj   (N x 128), K = 256
  k_gemm_bf16<256, false, true><<<dim3(N / 16, kC / 64), 128, 0, stream>>>(
      ybar, wtout, fm, N, kC, nullptr, feat);

  // layer_norm(fm) -> cat[:, :128]
  k_ln_bf16<<<rowGrd, rowBlk, 0, stream>>>(fm, ln_g, ln_b, catm, 2 * kC, 0, 0, N);

  // hpre = cat @ la_w1 + b1   (K = 256)
  k_gemm_bf16<256, true, false><<<dim3(N / 16, kC / 64), 128, 0, stream>>>(
      catm, wtla1, hpre, N, kC, la_b1, nullptr);

  // act2 = relu(layer_norm(hpre))
  k_ln_bf16<<<rowGrd, rowBlk, 0, stream>>>(hpre, la_ln_g, la_ln_b, act2, kC, 0, 1, N);

  // out = act2 @ la_w2 + b2   (K = 128)
  k_gemm_bf16<128, true, false><<<dim3(N / 16, kC / 64), 128, 0, stream>>>(
      act2, wtla2, out, N, kC, la_b2, nullptr);
}
